// Transolver_70815420776671
// MI455X (gfx1250) — matrix operand order repaired and checked by reading, not compiled, not run on hardware
//
#include <hip/hip_runtime.h>
#include <hip/hip_bf16.h>
#include <math.h>

#define BDIM 2
#define NTOK 16384
#define HIDD 512
#define HEADS 8
#define DHEAD 64
#define GSL 64
#define OUTD 4
#define BN (BDIM * NTOK)   // 32768 tokens
#define NKS 16             // split-K factor for slice aggregation

typedef __attribute__((ext_vector_type(16))) _Float16 v16h;
typedef __attribute__((ext_vector_type(8)))  _Float16 v8h;
typedef __attribute__((ext_vector_type(8)))  float    v8f;

__device__ __forceinline__ float gelu_f(float x) {
  return 0.5f * x * (1.0f + erff(x * 0.70710678118654752440f));
}
__device__ __forceinline__ float wave_sum(float v) {
#pragma unroll
  for (int o = 16; o > 0; o >>= 1) v += __shfl_xor(v, o, 32);
  return v;
}
__device__ __forceinline__ float wave_max(float v) {
#pragma unroll
  for (int o = 16; o > 0; o >>= 1) v = fmaxf(v, __shfl_xor(v, o, 32));
  return v;
}

// Fused epilogue for one 16x16 sub-tile; dual fp32/f16 store
__device__ __forceinline__ void store_tile(v8f acc, float* Cp, _Float16* Hp,
    const float* Rp, const float* bias, const float* scale, const float* shift,
    int act, int ms, int ns, int hi, int ml, int ldc, int ldr)
{
  const int n = ns + ml;
  const float bi = bias  ? bias[n]  : 0.0f;
  const float sc = scale ? scale[n] : 1.0f;
  const float sh = shift ? shift[n] : 0.0f;
#pragma unroll
  for (int r = 0; r < 8; ++r) {
    const int m = ms + hi * 8 + r;          // C/D layout: lanes 0-15 M=r, 16-31 M=8+r
    float v = acc[r] + bi;
    if (act) v = gelu_f(v);
    v = v * sc + sh;
    if (Rp) v += Rp[(long)m * ldr + n];
    if (Cp) Cp[(long)m * ldc + n] = v;
    if (Hp) Hp[(long)m * ldc + n] = (_Float16)v;
  }
}

// ---------------------------------------------------------------------------
// Contiguous-K f16 GEMM: C[m,n] = epi(sum_k A[m,k]*B[n,k]),  A,B f16, acc f32.
// Wave computes a 32x32 tile: 2 A-frags x 2 B-frags -> 4 v_wmma per k-step,
// pure f16 loads (no cvt in loop). Block = 8 waves as 4(M) x 2(N) -> 128x64.
// M%32==0, K%32==0, gridDim.x == Ncols/64. Batched via z = zh*zlo + zl.
// ---------------------------------------------------------------------------
__global__ __launch_bounds__(256) void gemm_h16(
    const _Float16* __restrict__ A, long a_off, long a_bh, long a_bl, int a_rs,
    const _Float16* __restrict__ Bw, long b_off, long b_bh, long b_bl, int b_rs,
    float* __restrict__ C, _Float16* __restrict__ Ch,
    long c_off, long c_bh, long c_bl, int ldc,
    const float* __restrict__ bias, int act,
    const float* __restrict__ scale, const float* __restrict__ shift,
    const float* __restrict__ resid, long r_off, int ldr,
    int M, int K, int zlo)
{
  const int lane = threadIdx.x & 31;
  const int wave = threadIdx.x >> 5;
  const int wm = wave >> 1, wn = wave & 1;
  const int m0 = blockIdx.y * 128 + wm * 32;
  const int n0 = blockIdx.x * 64 + wn * 32;
  if (m0 >= M) return;                       // wave-uniform
  const int z = blockIdx.z;
  const int zh = z / zlo, zl = z - zh * zlo;
  const int ml = lane & 15, hi = lane >> 4;
  const int akh = hi * 16;                   // A: lanes 0-15 k 0-15, lanes 16-31 k 16-31
  const int bkh = hi * 16;                   // B: lanes 0-15 k 0-15, lanes 16-31 k 16-31

  const _Float16* pa0 = A  + a_off + (long)zh*a_bh + (long)zl*a_bl + (long)(m0+ml)*a_rs + akh;
  const _Float16* pa1 = pa0 + (long)16 * a_rs;
  const _Float16* pb0 = Bw + b_off + (long)zh*b_bh + (long)zl*b_bl + (long)(n0+ml)*b_rs + bkh;
  const _Float16* pb1 = pb0 + (long)16 * b_rs;

  const v8f z8 = {0.f,0.f,0.f,0.f,0.f,0.f,0.f,0.f};
  v8f c00 = z8, c01 = z8, c10 = z8, c11 = z8;

  for (int k0 = 0; k0 < K; k0 += 32) {
    const v8h a0l = *(const v8h*)(pa0);
    const v8h a0h = *(const v8h*)(pa0 + 8);
    const v8h a1l = *(const v8h*)(pa1);
    const v8h a1h = *(const v8h*)(pa1 + 8);
    const v16h a0 = __builtin_shufflevector(a0l, a0h, 0,1,2,3,4,5,6,7,8,9,10,11,12,13,14,15);
    const v16h a1 = __builtin_shufflevector(a1l, a1h, 0,1,2,3,4,5,6,7,8,9,10,11,12,13,14,15);
    const v16h b0 = *(const v16h*)(pb0);
    const v16h b1 = *(const v16h*)(pb1);
    __builtin_prefetch(pa0 + 64, 0, 1);
    __builtin_prefetch(pb0 + 64, 0, 1);
    c00 = __builtin_amdgcn_wmma_f32_16x16x32_f16(false, a0, false, b0, (short)0, c00, false, false);
    c01 = __builtin_amdgcn_wmma_f32_16x16x32_f16(false, a0, false, b1, (short)0, c01, false, false);
    c10 = __builtin_amdgcn_wmma_f32_16x16x32_f16(false, a1, false, b0, (short)0, c10, false, false);
    c11 = __builtin_amdgcn_wmma_f32_16x16x32_f16(false, a1, false, b1, (short)0, c11, false, false);
    pa0 += 32; pa1 += 32; pb0 += 32; pb1 += 32;
  }

  const long cz = c_off + (long)zh * c_bh + (long)zl * c_bl;
  float*     Cp = C  ? C  + cz : nullptr;
  _Float16*  Hp = Ch ? Ch + cz : nullptr;
  const float* Rp = resid ? (resid + r_off) : nullptr;
  store_tile(c00, Cp, Hp, Rp, bias, scale, shift, act, m0,      n0,      hi, ml, ldc, ldr);
  store_tile(c01, Cp, Hp, Rp, bias, scale, shift, act, m0,      n0 + 16, hi, ml, ldc, ldr);
  store_tile(c10, Cp, Hp, Rp, bias, scale, shift, act, m0 + 16, n0,      hi, ml, ldc, ldr);
  store_tile(c11, Cp, Hp, Rp, bias, scale, shift, act, m0 + 16, n0 + 16, hi, ml, ldc, ldr);
}

// ---------------------------------------------------------------------------
// Strided "TN" f16 GEMM (both operands k-major), split-K fp32 partials:
//   Cpart[ks][m,n] = sum_{k in chunk} A[m*a_rs + k*a_cs] * B[n*b_rs + k*b_cs]
// Slice aggregation st_num = sw^T @ fx_feat (M=64, N=64, K=16384).
// ---------------------------------------------------------------------------
__global__ __launch_bounds__(256) void gemm_tn_h16(
    const _Float16* __restrict__ A, long a_off, long a_bh, long a_bl, long a_rs, long a_cs,
    const _Float16* __restrict__ Bw, long b_off, long b_bh, long b_bl, long b_rs, long b_cs,
    float* __restrict__ C, long c_off, long c_bh, long c_bl, int ldc, long c_ks,
    int M, int K, int zlo, int nks)
{
  const int lane = threadIdx.x & 31;
  const int wave = threadIdx.x >> 5;
  const int wm = wave >> 1, wn = wave & 1;
  const int m0 = wm * 32;
  const int n0 = blockIdx.x * 64 + wn * 32;
  if (m0 >= M) return;                       // wave-uniform
  const int ks = blockIdx.y;
  const int kchunk = K / nks;
  const int z = blockIdx.z;
  const int zh = z / zlo, zl = z - zh * zlo;
  const int ml = lane & 15, hi = lane >> 4;
  const int akh = hi * 16, bkh = hi * 16;

  const _Float16* pa0 = A + a_off + (long)zh*a_bh + (long)zl*a_bl
                      + (long)(m0+ml)*a_rs + ((long)ks*kchunk + akh)*a_cs;
  const _Float16* pa1 = pa0 + (long)16 * a_rs;
  const _Float16* pb0 = Bw + b_off + (long)zh*b_bh + (long)zl*b_bl
                      + (long)(n0+ml)*b_rs + ((long)ks*kchunk + bkh)*b_cs;
  const _Float16* pb1 = pb0 + (long)16 * b_rs;
  const long a_step = 32 * a_cs, b_step = 32 * b_cs;
  const long oa16 = 8 * a_cs;

  const v8f z8 = {0.f,0.f,0.f,0.f,0.f,0.f,0.f,0.f};
  v8f c00 = z8, c01 = z8, c10 = z8, c11 = z8;

  for (int k0 = 0; k0 < kchunk; k0 += 32) {
    v16h a0, a1, b0, b1;
#pragma unroll
    for (int e = 0; e < 8; ++e) {            // e*a_cs offsets are loop-invariant
      a0[e]     = pa0[(long)e * a_cs];
      a0[8 + e] = pa0[(long)e * a_cs + oa16];
      a1[e]     = pa1[(long)e * a_cs];
      a1[8 + e] = pa1[(long)e * a_cs + oa16];
    }
#pragma unroll
    for (int e = 0; e < 16; ++e) {
      b0[e] = pb0[(long)e * b_cs];
      b1[e] = pb1[(long)e * b_cs];
    }
    c00 = __builtin_amdgcn_wmma_f32_16x16x32_f16(false, a0, false, b0, (short)0, c00, false, false);
    c01 = __builtin_amdgcn_wmma_f32_16x16x32_f16(false, a0, false, b1, (short)0, c01, false, false);
    c10 = __builtin_amdgcn_wmma_f32_16x16x32_f16(false, a1, false, b0, (short)0, c10, false, false);
    c11 = __builtin_amdgcn_wmma_f32_16x16x32_f16(false, a1, false, b1, (short)0, c11, false, false);
    pa0 += a_step; pa1 += a_step; pb0 += b_step; pb1 += b_step;
  }

  float* Cp = C + c_off + (long)zh * c_bh + (long)zl * c_bl + (long)ks * c_ks;
  v8f accs[4] = {c00, c01, c10, c11};
#pragma unroll
  for (int mi = 0; mi < 2; ++mi)
#pragma unroll
    for (int ni = 0; ni < 2; ++ni) {
      const v8f acc = accs[mi * 2 + ni];
      const int n = n0 + ni * 16 + ml;
#pragma unroll
      for (int r = 0; r < 8; ++r) {
        const int m = m0 + mi * 16 + hi * 8 + r;
        Cp[(long)m * ldc + n] = acc[r];
      }
    }
}

// ---------------- weight fp32 -> f16 conversion ----------------------------
__global__ void f32_to_f16(const float* __restrict__ in, _Float16* __restrict__ out, long n)
{
  const long i = (long)blockIdx.x * 256 + threadIdx.x;
  if (i < n) out[i] = (_Float16)in[i];
}

// ---------------- LayerNorm over 512 features, one wave per token ----------
// outf (fp32) and/or outh (f16) may be null.
__global__ __launch_bounds__(256) void ln512(const float* __restrict__ in,
                                             float* __restrict__ outf,
                                             _Float16* __restrict__ outh,
                                             const float* __restrict__ g,
                                             const float* __restrict__ b)
{
  const int lane = threadIdx.x & 31, wave = threadIdx.x >> 5;
  const long tok = (long)blockIdx.x * 8 + wave;
  if (tok >= BN) return;
  const float4* p = (const float4*)(in + tok * HIDD + lane * 16);
  float x[16];
#pragma unroll
  for (int q = 0; q < 4; ++q) {
    float4 t = p[q];
    x[4*q] = t.x; x[4*q+1] = t.y; x[4*q+2] = t.z; x[4*q+3] = t.w;
  }
  float s = 0.f;
#pragma unroll
  for (int j = 0; j < 16; ++j) s += x[j];
  const float mean = wave_sum(s) * (1.0f / HIDD);
  float vs = 0.f;
#pragma unroll
  for (int j = 0; j < 16; ++j) { float d = x[j] - mean; vs += d * d; }
  const float rstd = rsqrtf(wave_sum(vs) * (1.0f / HIDD) + 1e-5f);
  float*    frow = outf ? outf + tok * HIDD + lane * 16 : nullptr;
  _Float16* hrow = outh ? outh + tok * HIDD + lane * 16 : nullptr;
#pragma unroll
  for (int j = 0; j < 16; ++j) {
    const int c = lane * 16 + j;
    const float v = (x[j] - mean) * rstd * g[c] + b[c];
    if (frow) frow[j] = v;
    if (hrow) hrow[j] = (_Float16)v;
  }
}

// ---------------- pre l1: concat(x,pos) (K=4) -> gelu; dual store -----------
__global__ __launch_bounds__(256) void pre_l1(const float* __restrict__ x,
                                              const float* __restrict__ pos,
                                              const float* __restrict__ w,
                                              const float* __restrict__ b,
                                              float* __restrict__ outf,
                                              _Float16* __restrict__ outh)
{
  const long i = (long)blockIdx.x * 256 + threadIdx.x;
  if (i >= (long)BN * 1024) return;
  const long m = i >> 10; const int o = (int)(i & 1023);
  const float f0 = x[m], f1 = pos[m*3], f2 = pos[m*3+1], f3 = pos[m*3+2];
  const float* wr = w + o * 4;
  const float v = gelu_f(wr[0]*f0 + wr[1]*f1 + wr[2]*f2 + wr[3]*f3 + b[o]);
  outf[i] = v;                 // residual for l2 (fp32)
  outh[i] = (_Float16)v;       // GEMM A operand (f16)
}

// ---------------- sb = sb2(gelu(sb1(pos[...,:2]))) --------------------------
__global__ __launch_bounds__(64) void sb_kernel(const float* __restrict__ pos,
    const float* __restrict__ w1, const float* __restrict__ b1,
    const float* __restrict__ w2, const float* __restrict__ b2,
    float* __restrict__ sb)
{
  __shared__ float hid[32];
  const long tok = blockIdx.x; const int t = threadIdx.x;
  const float p0 = pos[tok*3], p1 = pos[tok*3+1];
  if (t < 32) hid[t] = gelu_f(w1[t*2]*p0 + w1[t*2+1]*p1 + b1[t]);
  __syncthreads();
  float s = b2[t];
#pragma unroll
  for (int j = 0; j < 32; ++j) s += hid[j] * w2[t*32 + j];
  sb[tok*64 + t] = s;
}

// ------------- slice softmax: swh = softmax_g(logits/temp[h] + 0.1*sb) ------
__global__ __launch_bounds__(256) void softmax_slice(const float* __restrict__ logits,
    _Float16* __restrict__ swh,
    const float* __restrict__ temp, const float* __restrict__ sb)
{
  const int lane = threadIdx.x & 31, wave = threadIdx.x >> 5;
  const long row = (long)blockIdx.x * 8 + wave;   // row = z*NTOK + n, z=bb*8+h
  const long z = row / NTOK; const long n = row - z * NTOK;
  const int h = (int)(z & 7); const long bb = z >> 3;
  const float tinv = 1.0f / temp[h];
  const float* p = logits + row * 64;
  const float* s = sb + (bb * NTOK + n) * 64;
  float v0 = p[lane]      * tinv + 0.1f * s[lane];
  float v1 = p[lane + 32] * tinv + 0.1f * s[lane + 32];
  const float mx = wave_max(fmaxf(v0, v1));
  const float e0 = expf(v0 - mx), e1 = expf(v1 - mx);
  const float inv = 1.0f / wave_sum(e0 + e1);
  _Float16* q = swh + row * 64;
  q[lane] = (_Float16)(e0 * inv); q[lane + 32] = (_Float16)(e1 * inv);
}

// ------------- snorm[z,g] = sum_n sw[z,n,g]  (f16 input) --------------------
__global__ __launch_bounds__(256) void colsum64(const _Float16* __restrict__ sw,
                                                float* __restrict__ snorm)
{
  const int z = blockIdx.x;
  const int g = threadIdx.x & 63, j = threadIdx.x >> 6;
  const _Float16* p = sw + (long)z * NTOK * 64;
  float s = 0.f;
  for (int n = j; n < NTOK; n += 4) s += (float)p[(long)n * 64 + g];
  __shared__ float sh[256];
  sh[threadIdx.x] = s; __syncthreads();
  if (threadIdx.x < 64) snorm[z*64 + g] = sh[g] + sh[64+g] + sh[128+g] + sh[192+g];
}

// ------------- st = (sum of split-K partials) / (snorm + eps) --------------
__global__ void st_scale(const float* __restrict__ num,
                         const float* __restrict__ snorm, float* __restrict__ st)
{
  const int i = blockIdx.x * 256 + threadIdx.x;     // 16*64*64 = 65536
  if (i >= BDIM*HEADS*GSL*DHEAD) return;
  float s = 0.f;
#pragma unroll
  for (int p = 0; p < NKS; ++p) s += num[p * 65536 + i];
  st[i] = s / (snorm[i >> 6] + 1e-5f);
}

__global__ void kv_mean(const float* __restrict__ st, float* __restrict__ kv)
{
  const int i = blockIdx.x * 256 + threadIdx.x;     // B*G*D = 8192
  if (i >= BDIM*GSL*DHEAD) return;
  const int bb = i >> 12, rem = i & 4095;
  float s = 0.f;
#pragma unroll
  for (int h = 0; h < 8; ++h) s += st[((bb << 3) + h) * 4096 + rem];
  kv[i] = s * 0.125f;
}

// ------------- per-(b,h) 64x64 slice attention ------------------------------
__global__ __launch_bounds__(256) void slice_attn(const float* __restrict__ st,
    const float* __restrict__ kv, float* __restrict__ ost,
    const float* __restrict__ wq, const float* __restrict__ wk,
    const float* __restrict__ wv, const float* __restrict__ ascale,
    const float* __restrict__ srs_p)
{
  __shared__ float s_st[4096], s_kv[4096], s_q[4096], s_k[4096];
  const int z = blockIdx.x, bb = z >> 3, h = z & 7, t = threadIdx.x;
  for (int i = t; i < 4096; i += 256) {
    s_st[i] = st[(long)z * 4096 + i];
    s_kv[i] = kv[(long)bb * 4096 + i];
  }
  __syncthreads();
  for (int i = t; i < 4096; i += 256) {
    const int g = i >> 6, c = i & 63;
    float aq = 0.f, ak = 0.f;
    for (int d = 0; d < 64; ++d) {
      aq += s_st[g*64 + d] * wq[c*64 + d];
      ak += s_kv[g*64 + d] * wk[c*64 + d];
    }
    s_q[i] = aq; s_k[i] = ak;
  }
  __syncthreads();
  if (t < 128) {                                    // row-normalize q and k
    float* row = (t < 64) ? (s_q + t*64) : (s_k + (t - 64)*64);
    float ss = 0.f;
    for (int d = 0; d < 64; ++d) ss += row[d] * row[d];
    const float inv = 1.0f / fmaxf(sqrtf(ss), 1e-12f);
    for (int d = 0; d < 64; ++d) row[d] *= inv;
  }
  __syncthreads();
  if (t < 64) {
    const float asc = ascale[h], srs = srs_p[0];
    float lg[64], mx = -1e30f;
    for (int j = 0; j < 64; ++j) {
      float s = 0.f;
      for (int d = 0; d < 64; ++d) s += s_q[t*64 + d] * s_k[j*64 + d];
      s *= asc; lg[j] = s; mx = fmaxf(mx, s);
    }
    float sum = 0.f;
    for (int j = 0; j < 64; ++j) { lg[j] = expf(lg[j] - mx); sum += lg[j]; }
    const float inv = 1.0f / sum;
    float u[64];                                    // aw @ kv  (then @ Wv^T)
    for (int d = 0; d < 64; ++d) {
      float s = 0.f;
      for (int j = 0; j < 64; ++j) s += lg[j] * s_kv[j*64 + d];
      u[d] = s * inv;
    }
    for (int c = 0; c < 64; ++c) {
      float o = 0.f;
      for (int d = 0; d < 64; ++d) o += u[d] * wv[c*64 + d];
      ost[(long)z * 4096 + t*64 + c] = o + srs * s_st[t*64 + c];
    }
  }
}

// ------------- transpose ost (64x64 per z) to f16, k-contiguous for WMMA ----
__global__ void ost_transpose(const float* __restrict__ ost, _Float16* __restrict__ ostt)
{
  const int i = blockIdx.x * 256 + threadIdx.x;     // 65536
  if (i >= BDIM*HEADS*GSL*DHEAD) return;
  const int z = i >> 12, r = i & 4095, g = r >> 6, c = r & 63;
  ostt[(z << 12) + (c << 6) + g] = (_Float16)ost[i];
}

// ------------- SE: token mean, tiny MLP, scale ------------------------------
__global__ __launch_bounds__(256) void se_mean(const float* __restrict__ fx,
                                               float* __restrict__ sevec)
{
  const int blk = blockIdx.x;                       // B*HIDD = 1024
  const int bb = blk >> 9, c = blk & 511;
  const float* p = fx + (long)bb * NTOK * HIDD + c;
  float s = 0.f;
  for (int n = threadIdx.x; n < NTOK; n += 256) s += p[(long)n * HIDD];
  __shared__ float sh[256];
  sh[threadIdx.x] = s; __syncthreads();
  for (int o = 128; o > 0; o >>= 1) {
    if (threadIdx.x < o) sh[threadIdx.x] += sh[threadIdx.x + o];
    __syncthreads();
  }
  if (threadIdx.x == 0) sevec[blk] = sh[0] * (1.0f / NTOK);
}

__global__ __launch_bounds__(512) void se_mlp(const float* __restrict__ sevec,
    const float* __restrict__ w1, const float* __restrict__ b1,
    const float* __restrict__ w2, const float* __restrict__ b2,
    float* __restrict__ seout)
{
  __shared__ float hid[128];
  const int bb = blockIdx.x, t = threadIdx.x;
  const float* sv = sevec + bb * HIDD;
  if (t < 128) {
    float s = b1[t];
    for (int c = 0; c < HIDD; ++c) s += sv[c] * w1[t*HIDD + c];
    hid[t] = gelu_f(s);
  }
  __syncthreads();
  float s = b2[t];
  for (int j = 0; j < 128; ++j) s += hid[j] * w2[t*128 + j];
  seout[bb * HIDD + t] = 1.0f / (1.0f + expf(-s));
}

__global__ void se_apply(float* __restrict__ fx, const float* __restrict__ seout)
{
  const long i = (long)blockIdx.x * 256 + threadIdx.x;
  if (i >= (long)BN * HIDD) return;
  const long tok = i >> 9; const long bb = tok / NTOK; const int c = (int)(i & 511);
  fx[i] *= seout[bb * HIDD + c];
}

// ------------- final heads: h2(gelu-h1-out) + skip(fx) ----------------------
__global__ void head_kernel(const float* __restrict__ h1out,
    const float* __restrict__ fx,
    const float* __restrict__ h2w, const float* __restrict__ h2b,
    const float* __restrict__ skw, const float* __restrict__ skb,
    float* __restrict__ out)
{
  const long i = (long)blockIdx.x * 256 + threadIdx.x;   // BN*4
  if (i >= (long)BN * OUTD) return;
  const long m = i >> 2; const int o = (int)(i & 3);
  const float* a  = h1out + m * HIDD;
  const float* b  = fx + m * HIDD;
  const float* w1 = h2w + o * HIDD;
  const float* w2 = skw + o * HIDD;
  float s = h2b[o] + skb[o];
  for (int k = 0; k < HIDD; ++k) s += a[k] * w1[k] + b[k] * w2[k];
  out[i] = s;
}

// ===========================================================================
extern "C" void kernel_launch(void* const* d_in, const int* in_sizes, int n_in,
                              void* d_out, int out_size, void* d_ws, size_t ws_size,
                              hipStream_t stream)
{
  (void)in_sizes; (void)n_in; (void)out_size; (void)ws_size;
  const float* x   = (const float*)d_in[0];
  const float* pos = (const float*)d_in[1];
  auto P = [&](int i) { return (const float*)d_in[i]; };

  // workspace carve-up (256B-aligned)
  char* ws = (char*)d_ws;
  size_t off = 0;
  auto carve = [&](size_t bytes) -> void* {
    void* p = ws + off; off += (bytes + 255) & ~(size_t)255; return p;
  };
  float*    W0  = (float*)   carve((size_t)BN * 1024 * 4);  // l1 out fp32 (l2 residual)
  _Float16* W0H = (_Float16*)carve((size_t)BN * 1024 * 2);  // l1 out f16 / m1 hidden
  _Float16* W1H = (_Float16*)carve((size_t)BN * 1024 * 2);  // l2 out f16
  float*    FX  = (float*)   carve((size_t)BN * HIDD * 4);  // running features fp32
  float*    T1  = (float*)   carve((size_t)BN * HIDD * 4);  // pre-LN temp fp32
  _Float16* T1H = (_Float16*)carve((size_t)BN * HIDD * 2);  // LN out f16 (GEMM A)
  _Float16* XMH = (_Float16*)carve((size_t)BN * HIDD * 2);  // px out / ox  f16
  _Float16* FFH = (_Float16*)carve((size_t)BN * HIDD * 2);  // pfx out f16
  float*    FF  = (float*)   carve((size_t)BN * HIDD * 4);  // h1 out fp32
  float*    SW  = (float*)   carve((size_t)BN * HIDD * 4);  // slice logits fp32
  _Float16* SWH = (_Float16*)carve((size_t)BN * HIDD * 2);  // slice softmax f16
  float*    SB  = (float*)   carve((size_t)BN * 64 * 4);
  float* SNORM = (float*)carve(4096 * 4);
  float* STP   = (float*)carve((size_t)NKS * 65536 * 4);    // split-K partials
  float* ST    = (float*)carve(65536 * 4);
  float* KV    = (float*)carve(8192 * 4);
  float* OST   = (float*)carve(65536 * 4);
  float* SEV   = (float*)carve(1024 * 4);
  float* SEO   = (float*)carve(1024 * 4);
  _Float16* HL2  = (_Float16*)carve(1048576 * 2);
  _Float16* HL3  = (_Float16*)carve(524288 * 2);
  _Float16* HH1  = (_Float16*)carve(262144 * 2);
  _Float16* HPX  = (_Float16*)carve(262144 * 2);
  _Float16* HPFX = (_Float16*)carve(262144 * 2);
  _Float16* HPSL = (_Float16*)carve(4096 * 2);
  _Float16* HOUT = (_Float16*)carve(262144 * 2);
  _Float16* HM1  = (_Float16*)carve(524288 * 2);
  _Float16* HM2  = (_Float16*)carve(524288 * 2);
  _Float16* OSTT = (_Float16*)carve(65536 * 2);

  const dim3 TB(256);
  const long NG = (long)NTOK * 64;
  auto cvt = [&](const float* src, _Float16* dst, long n) {
    f32_to_f16<<<(int)((n + 255) / 256), TB, 0, stream>>>(src, dst, n);
  };

  // weights used by WMMA GEMMs -> f16 (once per launch, ~9M elements)
  cvt(P(4),   HL2, 1048576);
  cvt(P(6),   HL3, 524288);
  cvt(P(148), HH1, 262144);

  // ---- pre-MLP ----
  pre_l1<<<(BN * 1024 + 255) / 256, TB, 0, stream>>>(x, pos, P(2), P(3), W0, W0H);
  gemm_h16<<<dim3(16, BN/128, 1), TB, 0, stream>>>(               // l2 +gelu +resid -> f16
      W0H, 0,0,0, 1024,  HL2, 0,0,0, 1024,  nullptr, W1H, 0,0,0, 1024,
      P(5), 1, nullptr, nullptr, W0, 0, 1024, BN, 1024, 1);
  gemm_h16<<<dim3(8, BN/128, 1), TB, 0, stream>>>(                // l3 *scale+shift -> fp32
      W1H, 0,0,0, 1024,  HL3, 0,0,0, 1024,  FX, nullptr, 0,0,0, HIDD,
      P(7), 0, P(8), P(9), nullptr, 0, 0, BN, 1024, 1);

  // ---- 4 Transolver blocks ----
  for (int blk = 0; blk < 4; ++blk) {
    const int pb = 10 + 34 * blk;
    const float *ln1g=P(pb+0),*ln1b=P(pb+1),*ln1pg=P(pb+2),*ln1pb=P(pb+3);
    const float *ln2g=P(pb+4),*ln2b=P(pb+5),*ln2pg=P(pb+6),*ln2pb=P(pb+7);
    const float *pxb=P(pb+9),*pfxb=P(pb+11);
    const float *pslb=P(pb+13),*temp=P(pb+14);
    const float *wq=P(pb+15),*wk=P(pb+16),*wv=P(pb+17),*srs=P(pb+18);
    const float *outb=P(pb+20),*ascale=P(pb+21);
    const float *m1b=P(pb+23),*m2b=P(pb+25);
    const float *sb1w=P(pb+26),*sb1b=P(pb+27),*sb2w=P(pb+28),*sb2b=P(pb+29);
    const float *se1w=P(pb+30),*se1b=P(pb+31),*se2w=P(pb+32),*se2b=P(pb+33);

    cvt(P(pb+8),  HPX,  262144);  cvt(P(pb+10), HPFX, 262144);
    cvt(P(pb+12), HPSL, 4096);    cvt(P(pb+19), HOUT, 262144);
    cvt(P(pb+22), HM1,  524288);  cvt(P(pb+24), HM2,  524288);

    sb_kernel<<<BN, 64, 0, stream>>>(pos, sb1w, sb1b, sb2w, sb2b, SB);
    ln512<<<BN/8, TB, 0, stream>>>(FX, nullptr, T1H, ln1g, ln1b);
    gemm_h16<<<dim3(8, BN/128, 1), TB, 0, stream>>>(              // pfx -> f16
        T1H, 0,0,0, HIDD,  HPFX, 0,0,0, HIDD,  nullptr, FFH, 0,0,0, HIDD,
        pfxb, 0, nullptr, nullptr, nullptr, 0, 0, BN, HIDD, 1);
    gemm_h16<<<dim3(8, BN/128, 1), TB, 0, stream>>>(              // px -> f16
        T1H, 0,0,0, HIDD,  HPX, 0,0,0, HIDD,  nullptr, XMH, 0,0,0, HIDD,
        pxb, 0, nullptr, nullptr, nullptr, 0, 0, BN, HIDD, 1);
    // slice logits per (b,h): A = xm head slice (f16), B = pslice.w (f16)
    gemm_h16<<<dim3(1, NTOK/128, 16), TB, 0, stream>>>(
        XMH, 0, (long)NTOK*HIDD, 64, HIDD,
        HPSL, 0, 0, 0, 64,
        SW, nullptr, 0, (long)HEADS*NG, NG, 64,
        pslb, 0, nullptr, nullptr, nullptr, 0, 0, NTOK, 64, 8);
    softmax_slice<<<(BDIM*HEADS*NTOK)/8, TB, 0, stream>>>(SW, SWH, temp, SB);
    colsum64<<<BDIM*HEADS, TB, 0, stream>>>(SWH, SNORM);
    // st numerator: sw^T @ fx_feat  (M=64, N=64, K=16384, split-K partials)
    gemm_tn_h16<<<dim3(1, NKS, 16), TB, 0, stream>>>(
        SWH, 0, (long)HEADS*NG, NG, 1, 64,
        FFH, 0, (long)NTOK*HIDD, 64, 1, HIDD,
        STP, 0, 32768, 4096, 64, 65536,
        64, NTOK, 8, NKS);
    st_scale<<<256, TB, 0, stream>>>(STP, SNORM, ST);
    kv_mean<<<32, TB, 0, stream>>>(ST, KV);
    slice_attn<<<BDIM*HEADS, TB, 0, stream>>>(ST, KV, OST, wq, wk, wv, ascale, srs);
    ost_transpose<<<256, TB, 0, stream>>>(OST, OSTT);
    // de-slice: ox = sw @ ost^T  -> head columns of XMH (f16)
    gemm_h16<<<dim3(1, NTOK/128, 16), TB, 0, stream>>>(
        SWH, 0, (long)HEADS*NG, NG, 64,
        OSTT, 0, 32768, 4096, 64,
        nullptr, XMH, 0, (long)NTOK*HIDD, 64, HIDD,
        nullptr, 0, nullptr, nullptr, nullptr, 0, 0, NTOK, 64, 8);
    // out projection + residual fx -> fp32 (pre-LN)
    gemm_h16<<<dim3(8, BN/128, 1), TB, 0, stream>>>(
        XMH, 0,0,0, HIDD,  HOUT, 0,0,0, HIDD,  T1, nullptr, 0,0,0, HIDD,
        outb, 0, nullptr, nullptr, FX, 0, HIDD, BN, HIDD, 1);
    ln512<<<BN/8, TB, 0, stream>>>(T1, FX, nullptr, ln1pg, ln1pb);
    // MLP
    ln512<<<BN/8, TB, 0, stream>>>(FX, nullptr, T1H, ln2g, ln2b);
    gemm_h16<<<dim3(16, BN/128, 1), TB, 0, stream>>>(             // m1 + gelu -> f16
        T1H, 0,0,0, HIDD,  HM1, 0,0,0, HIDD,  nullptr, W0H, 0,0,0, 1024,
        m1b, 1, nullptr, nullptr, nullptr, 0, 0, BN, HIDD, 1);
    gemm_h16<<<dim3(8, BN/128, 1), TB, 0, stream>>>(              // m2 + resid -> fp32
        W0H, 0,0,0, 1024,  HM2, 0,0,0, 1024,  T1, nullptr, 0,0,0, HIDD,
        m2b, 0, nullptr, nullptr, FX, 0, HIDD, BN, 1024, 1);
    ln512<<<BN/8, TB, 0, stream>>>(T1, FX, nullptr, ln2pg, ln2pb);
    // SE gate
    se_mean<<<BDIM*HIDD, TB, 0, stream>>>(FX, SEV);
    se_mlp<<<BDIM, 512, 0, stream>>>(SEV, se1w, se1b, se2w, se2b, SEO);
    se_apply<<<((long)BN*HIDD + 255) / 256, TB, 0, stream>>>(FX, SEO);
  }

  // ---- head ----
  ln512<<<BN/8, TB, 0, stream>>>(FX, nullptr, T1H, P(146), P(147));
  gemm_h16<<<dim3(8, BN/128, 1), TB, 0, stream>>>(                // h1 + gelu -> fp32
      T1H, 0,0,0, HIDD,  HH1, 0,0,0, HIDD,  FF, nullptr, 0,0,0, HIDD,
      P(149), 1, nullptr, nullptr, nullptr, 0, 0, BN, HIDD, 1);
  head_kernel<<<(BN*OUTD + 255) / 256, TB, 0, stream>>>(
      FF, FX, P(150), P(151), P(152), P(153), (float*)d_out);
}
